// MultiHeadAttention_26585847562509
// MI455X (gfx1250) — matrix operand order repaired and checked by reading, not compiled, not run on hardware
//
#include <hip/hip_runtime.h>

// Problem constants (match reference)
#define DMODEL 1024
#define NHEADS 16
#define DKH    64
#define SEQ    2048
#define NBATCH 2
#define MROWS  (NBATCH * SEQ)   // 4096 token rows

// CDNA5 WMMA vector types (probe-confirmed signatures)
typedef __attribute__((ext_vector_type(16))) __bf16 v16bf;
typedef __attribute__((ext_vector_type(8)))  __bf16 v8bf;
typedef __attribute__((ext_vector_type(4)))  __bf16 v4bf;
typedef __attribute__((ext_vector_type(8)))  float  v8f;
typedef __attribute__((ext_vector_type(4)))  float  v4f;

// ===========================================================================
// Split-precision scheme: x = hi + lo (both bf16). GEMMs accumulate
// hi*hi + hi*lo + lo*hi in fp32 via v_wmma_f32_16x16x32_bf16 -> ~fp32
// accuracy at 3 WMMAs per K=32 chunk. Splits are precomputed ONCE per
// element (not per tile) so the GEMM inner loops are pure load + WMMA.
// ===========================================================================

// ---------------------------------------------------------------------------
// Pre-split kernel: fp32 -> (hi, lo) bf16 arrays. n is a multiple of 1024.
// ---------------------------------------------------------------------------
__global__ __launch_bounds__(256)
void split_f32(const float* __restrict__ src, __bf16* __restrict__ hi,
               __bf16* __restrict__ lo, int n) {
    const int i = (blockIdx.x * 256 + threadIdx.x) * 4;
    if (i >= n) return;
    v4f x = *(const v4f*)(src + i);
    v4bf h, l;
#pragma unroll
    for (int j = 0; j < 4; ++j) {
        __bf16 hj = (__bf16)x[j];
        h[j] = hj;
        l[j] = (__bf16)(x[j] - (float)hj);
    }
    *(v4bf*)(hi + i) = h;
    *(v4bf*)(lo + i) = l;
}

// ---------------------------------------------------------------------------
// Fragment loaders (ISA 7.12.2 layouts), operands already bf16.
// A 16x32 (MxK): lanes 0-15: M=lane, K={0..7,16..23}; lanes 16-31: +8 on K.
// B 32x16 (KxN): lanes 0-15: N=lane, K={0..7,16..23}; lanes 16-31: +8 on K.
// ---------------------------------------------------------------------------
__device__ __forceinline__ v16bf load_a_bf(const __bf16* __restrict__ base,
                                           int lda, int lane) {
    const __bf16* p = base + (size_t)(lane & 15) * lda + ((lane >> 4) << 3);
    v8bf a0 = *(const v8bf*)(p);
    v8bf a1 = *(const v8bf*)(p + 16);
    return __builtin_shufflevector(a0, a1, 0, 1, 2, 3, 4, 5, 6, 7,
                                   8, 9, 10, 11, 12, 13, 14, 15);
}

__device__ __forceinline__ v16bf load_b_bf(const __bf16* __restrict__ base,
                                           int ldb, int lane) {
    const __bf16* p = base + (size_t)(lane & 15) * ldb + ((lane >> 4) << 3);
    v8bf b0 = *(const v8bf*)(p);
    v8bf b1 = *(const v8bf*)(p + 16);
    return __builtin_shufflevector(b0, b1, 0, 1, 2, 3, 4, 5, 6, 7,
                                   8, 9, 10, 11, 12, 13, 14, 15);
}

// fp32 A fragment with on-the-fly split (used only for the attn matrix,
// where each element is consumed exactly once).
__device__ __forceinline__ void load_a_f32(const float* __restrict__ base,
                                           int lda, int lane,
                                           v16bf& ah, v16bf& al) {
    const float* p = base + (size_t)(lane & 15) * lda + ((lane >> 4) << 3);
    v8f x0 = *(const v8f*)(p);
    v8f x1 = *(const v8f*)(p + 16);
#pragma unroll
    for (int j = 0; j < 8; ++j) {
        __bf16 h0 = (__bf16)x0[j];
        __bf16 h1 = (__bf16)x1[j];
        ah[j]     = h0;  al[j]     = (__bf16)(x0[j] - (float)h0);
        ah[j + 8] = h1;  al[j + 8] = (__bf16)(x1[j] - (float)h1);
    }
}

// ---------------------------------------------------------------------------
// Wave strip: C(16 x 16*NT) = A(16xKd) * B(NT*16 x Kd)^T, bf16x3.
// WMMAs are grouped so consecutive instructions hit different accumulators
// (independent -> fewer hazard NOPs, better XDL pipelining).
// ---------------------------------------------------------------------------
template <int NT>
__device__ __forceinline__ void gemm_strip_bf(const __bf16* __restrict__ Ah,
                                              const __bf16* __restrict__ Al, int lda,
                                              const __bf16* __restrict__ Bh,
                                              const __bf16* __restrict__ Bl, int ldb,
                                              int Kd, int lane, v8f acc[NT]) {
    v8f zero = {};
#pragma unroll
    for (int t = 0; t < NT; ++t) acc[t] = zero;
    for (int k0 = 0; k0 < Kd; k0 += 32) {
        v16bf ah = load_a_bf(Ah + k0, lda, lane);
        v16bf al = load_a_bf(Al + k0, lda, lane);
        v16bf bh[NT], bl[NT];
#pragma unroll
        for (int t = 0; t < NT; ++t) {
            bh[t] = load_b_bf(Bh + (size_t)(t * 16) * ldb + k0, ldb, lane);
            bl[t] = load_b_bf(Bl + (size_t)(t * 16) * ldb + k0, ldb, lane);
        }
#pragma unroll
        for (int t = 0; t < NT; ++t)
            acc[t] = __builtin_amdgcn_wmma_f32_16x16x32_bf16(false, ah, false, bh[t], (short)0, acc[t], false, false);
#pragma unroll
        for (int t = 0; t < NT; ++t)
            acc[t] = __builtin_amdgcn_wmma_f32_16x16x32_bf16(false, ah, false, bl[t], (short)0, acc[t], false, false);
#pragma unroll
        for (int t = 0; t < NT; ++t)
            acc[t] = __builtin_amdgcn_wmma_f32_16x16x32_bf16(false, al, false, bh[t], (short)0, acc[t], false, false);
    }
}

// Same, but A is fp32 split on the fly (attn @ V).
template <int NT>
__device__ __forceinline__ void gemm_strip_f32a(const float* __restrict__ A, int lda,
                                                const __bf16* __restrict__ Bh,
                                                const __bf16* __restrict__ Bl, int ldb,
                                                int Kd, int lane, v8f acc[NT]) {
    v8f zero = {};
#pragma unroll
    for (int t = 0; t < NT; ++t) acc[t] = zero;
    for (int k0 = 0; k0 < Kd; k0 += 32) {
        v16bf ah, al;
        load_a_f32(A + k0, lda, lane, ah, al);
        v16bf bh[NT], bl[NT];
#pragma unroll
        for (int t = 0; t < NT; ++t) {
            bh[t] = load_b_bf(Bh + (size_t)(t * 16) * ldb + k0, ldb, lane);
            bl[t] = load_b_bf(Bl + (size_t)(t * 16) * ldb + k0, ldb, lane);
        }
#pragma unroll
        for (int t = 0; t < NT; ++t)
            acc[t] = __builtin_amdgcn_wmma_f32_16x16x32_bf16(false, ah, false, bh[t], (short)0, acc[t], false, false);
#pragma unroll
        for (int t = 0; t < NT; ++t)
            acc[t] = __builtin_amdgcn_wmma_f32_16x16x32_bf16(false, ah, false, bl[t], (short)0, acc[t], false, false);
#pragma unroll
        for (int t = 0; t < NT; ++t)
            acc[t] = __builtin_amdgcn_wmma_f32_16x16x32_bf16(false, al, false, bh[t], (short)0, acc[t], false, false);
    }
}

// ---------------------------------------------------------------------------
// Projection GEMM: Out[m,n] = sum_k X[m,k]*W[n,k] + bias[n]
// M=4096, N=1024, Kd=1024. One wave -> 16x64 tile; grid.x = 512.
// mode 0: write split bf16, layout [m][n]           (Q, K projections)
// mode 1: write split bf16, per-head transposed V:  Vt[((b*16+h)*64+d)*2048+s]
// mode 2: write fp32 to OutF                        (final output projection)
// ---------------------------------------------------------------------------
__global__ __launch_bounds__(256)
void proj_gemm(const __bf16* __restrict__ Xh, const __bf16* __restrict__ Xl,
               const __bf16* __restrict__ Wh, const __bf16* __restrict__ Wl,
               const float* __restrict__ bias,
               float* __restrict__ OutF,
               __bf16* __restrict__ OutH, __bf16* __restrict__ OutL,
               int mode) {
    const int lane = threadIdx.x & 31;
    const int wid  = blockIdx.x * 8 + (threadIdx.x >> 5);
    const int strips = DMODEL / 64;                 // 16
    const int m0 = (wid / strips) * 16;
    const int n0 = (wid % strips) * 64;

    v8f acc[4];
    gemm_strip_bf<4>(Xh + (size_t)m0 * DMODEL, Xl + (size_t)m0 * DMODEL, DMODEL,
                     Wh + (size_t)n0 * DMODEL, Wl + (size_t)n0 * DMODEL, DMODEL,
                     DMODEL, lane, acc);

    const int hi = lane >> 4;
    const int nl = lane & 15;
#pragma unroll
    for (int t = 0; t < 4; ++t) {
        const int n = n0 + t * 16 + nl;
        const float bv = bias[n];
#pragma unroll
        for (int r = 0; r < 8; ++r) {
            const int m = m0 + r + 8 * hi;          // C layout: VGPR r, lane halves
            const float val = acc[t][r] + bv;
            if (mode == 2) {
                OutF[(size_t)m * DMODEL + n] = val;
            } else {
                const __bf16 vh = (__bf16)val;
                const __bf16 vl = (__bf16)(val - (float)vh);
                size_t idx;
                if (mode == 1) {
                    const int b = m >> 11, s = m & (SEQ - 1);
                    const int h = n >> 6,  d = n & 63;
                    idx = (((size_t)(b * NHEADS + h) * DKH + d) * SEQ) + s;
                } else {
                    idx = (size_t)m * DMODEL + n;
                }
                OutH[idx] = vh;
                OutL[idx] = vl;
            }
        }
    }
}

// ---------------------------------------------------------------------------
// Scores: attn_raw[b,h,q,k] = (1/8) * sum_d Q[b,q,h*64+d]*K[b,k,h*64+d]
// Per (b,h): M=N=2048, Kd=64, lda=ldb=1024 (bf16 split).  grid = (512,1,32)
// ---------------------------------------------------------------------------
__global__ __launch_bounds__(256)
void attn_scores(const __bf16* __restrict__ Qh, const __bf16* __restrict__ Ql,
                 const __bf16* __restrict__ Kh, const __bf16* __restrict__ Kl,
                 float* __restrict__ attn) {
    const int lane = threadIdx.x & 31;
    const int bh = blockIdx.z;
    const int b  = bh >> 4, h = bh & 15;
    const int wid = blockIdx.x * 8 + (threadIdx.x >> 5);
    const int strips = SEQ / 64;                    // 32
    const int m0 = (wid / strips) * 16;
    const int n0 = (wid % strips) * 64;

    const size_t aoff = (size_t)b * SEQ * DMODEL + (size_t)m0 * DMODEL + h * DKH;
    const size_t boff = (size_t)b * SEQ * DMODEL + (size_t)n0 * DMODEL + h * DKH;

    v8f acc[4];
    gemm_strip_bf<4>(Qh + aoff, Ql + aoff, DMODEL,
                     Kh + boff, Kl + boff, DMODEL, DKH, lane, acc);

    float* Ob = attn + (size_t)bh * SEQ * SEQ;
    const int hi = lane >> 4;
    const int nl = lane & 15;
    const float scale = 0.125f;                     // 1/sqrt(64)
#pragma unroll
    for (int t = 0; t < 4; ++t)
#pragma unroll
        for (int r = 0; r < 8; ++r)
            Ob[(size_t)(m0 + r + 8 * hi) * SEQ + (n0 + t * 16 + nl)] = acc[t][r] * scale;
}

// ---------------------------------------------------------------------------
// In-place row softmax over attn rows of length 2048.
// One 256-thread block per row; grid.x = 2*16*2048 = 65536.
// ---------------------------------------------------------------------------
__global__ __launch_bounds__(256)
void softmax_rows(float* __restrict__ attn) {
    __shared__ float red[256];
    float* p = attn + (size_t)blockIdx.x * SEQ;
    const int tid = threadIdx.x;

    float v[8];
    float mx = -3.402823466e38f;
#pragma unroll
    for (int j = 0; j < 8; ++j) { v[j] = p[tid + 256 * j]; mx = fmaxf(mx, v[j]); }
    red[tid] = mx;
    __syncthreads();
    for (int off = 128; off > 0; off >>= 1) {
        if (tid < off) red[tid] = fmaxf(red[tid], red[tid + off]);
        __syncthreads();
    }
    mx = red[0];
    __syncthreads();

    float sum = 0.f;
#pragma unroll
    for (int j = 0; j < 8; ++j) { v[j] = expf(v[j] - mx); sum += v[j]; }
    red[tid] = sum;
    __syncthreads();
    for (int off = 128; off > 0; off >>= 1) {
        if (tid < off) red[tid] += red[tid + off];
        __syncthreads();
    }
    const float inv = 1.0f / red[0];
#pragma unroll
    for (int j = 0; j < 8; ++j) p[tid + 256 * j] = v[j] * inv;
}

// ---------------------------------------------------------------------------
// Context: ctx = attn @ V, via per-head transposed split V.
// Per (b,h): M=2048, N=64, Kd=2048. Emits ctx pre-split for the out-proj.
// grid = (16,1,32)
// ---------------------------------------------------------------------------
__global__ __launch_bounds__(256)
void attn_context(const float* __restrict__ attn,
                  const __bf16* __restrict__ VtH, const __bf16* __restrict__ VtL,
                  __bf16* __restrict__ ctxH, __bf16* __restrict__ ctxL) {
    const int lane = threadIdx.x & 31;
    const int bh = blockIdx.z;
    const int b  = bh >> 4, h = bh & 15;
    const int wid = blockIdx.x * 8 + (threadIdx.x >> 5);   // 0..127
    const int m0 = wid * 16;

    const float* Ab = attn + (size_t)bh * SEQ * SEQ + (size_t)m0 * SEQ;
    const size_t boff = (size_t)bh * DKH * SEQ;            // [64 x 2048] per head

    v8f acc[4];
    gemm_strip_f32a<4>(Ab, SEQ, VtH + boff, VtL + boff, SEQ, SEQ, lane, acc);

    const int hi = lane >> 4;
    const int nl = lane & 15;
#pragma unroll
    for (int t = 0; t < 4; ++t)
#pragma unroll
        for (int r = 0; r < 8; ++r) {
            const int m = m0 + r + 8 * hi;
            const int col = h * DKH + t * 16 + nl;
            const size_t idx = (size_t)(b * SEQ + m) * DMODEL + col;
            const float val = acc[t][r];
            const __bf16 vh = (__bf16)val;
            ctxH[idx] = vh;
            ctxL[idx] = (__bf16)(val - (float)vh);
        }
}

// ---------------------------------------------------------------------------
// Host orchestration.  Workspace (bf16 hi/lo arrays, ~128 MB):
//  4 weights (hi+lo)  : 8 x 2 MB
//  3 inputs  (hi+lo)  : 6 x 8 MB
//  Qp,Kp     (hi+lo)  : 4 x 8 MB
//  Vt        (hi+lo)  : 2 x 8 MB
//  ctx       (hi+lo)  : 2 x 8 MB
// d_out: output [2,2048,1024] fp32, then attn [2,16,2048,2048] fp32.
// ---------------------------------------------------------------------------
extern "C" void kernel_launch(void* const* d_in, const int* in_sizes, int n_in,
                              void* d_out, int out_size, void* d_ws, size_t ws_size,
                              hipStream_t stream) {
    const float* query = (const float*)d_in[0];
    const float* key_  = (const float*)d_in[1];
    const float* value = (const float*)d_in[2];
    const float* Wq = (const float*)d_in[3];
    const float* bq = (const float*)d_in[4];
    const float* Wk = (const float*)d_in[5];
    const float* bk = (const float*)d_in[6];
    const float* Wv = (const float*)d_in[7];
    const float* bv = (const float*)d_in[8];
    const float* Wo = (const float*)d_in[9];
    const float* bo = (const float*)d_in[10];

    float* out  = (float*)d_out;
    float* attn = out + (size_t)MROWS * DMODEL;

    const size_t NW = (size_t)DMODEL * DMODEL;  // 1 Mi elements per weight
    const size_t NX = (size_t)MROWS * DMODEL;   // 4 Mi elements per activation

    __bf16* p = (__bf16*)d_ws;
    __bf16 *WqH = p, *WqL = p + NW;            p += 2 * NW;
    __bf16 *WkH = p, *WkL = p + NW;            p += 2 * NW;
    __bf16 *WvH = p, *WvL = p + NW;            p += 2 * NW;
    __bf16 *WoH = p, *WoL = p + NW;            p += 2 * NW;
    __bf16 *qH  = p, *qL  = p + NX;            p += 2 * NX;
    __bf16 *kH  = p, *kL  = p + NX;            p += 2 * NX;
    __bf16 *vH  = p, *vL  = p + NX;            p += 2 * NX;
    __bf16 *QpH = p, *QpL = p + NX;            p += 2 * NX;
    __bf16 *KpH = p, *KpL = p + NX;            p += 2 * NX;
    __bf16 *VtH = p, *VtL = p + NX;            p += 2 * NX;
    __bf16 *cxH = p, *cxL = p + NX;            p += 2 * NX;

    const dim3 blk(256, 1, 1);

    // One-time fp32 -> bf16 hi/lo splits (1 cvt pair per element).
    split_f32<<<(int)(NW / 1024), blk, 0, stream>>>(Wq, WqH, WqL, (int)NW);
    split_f32<<<(int)(NW / 1024), blk, 0, stream>>>(Wk, WkH, WkL, (int)NW);
    split_f32<<<(int)(NW / 1024), blk, 0, stream>>>(Wv, WvH, WvL, (int)NW);
    split_f32<<<(int)(NW / 1024), blk, 0, stream>>>(Wo, WoH, WoL, (int)NW);
    split_f32<<<(int)(NX / 1024), blk, 0, stream>>>(query, qH, qL, (int)NX);
    split_f32<<<(int)(NX / 1024), blk, 0, stream>>>(key_,  kH, kL, (int)NX);
    split_f32<<<(int)(NX / 1024), blk, 0, stream>>>(value, vH, vL, (int)NX);

    // Q/K/V projections (outputs emitted pre-split; V transposed per head).
    proj_gemm<<<512, blk, 0, stream>>>(qH, qL, WqH, WqL, bq, nullptr, QpH, QpL, 0);
    proj_gemm<<<512, blk, 0, stream>>>(kH, kL, WkH, WkL, bk, nullptr, KpH, KpL, 0);
    proj_gemm<<<512, blk, 0, stream>>>(vH, vL, WvH, WvL, bv, nullptr, VtH, VtL, 1);

    // Attention scores -> softmax (fp32, straight into d_out attn region).
    attn_scores<<<dim3(512, 1, NBATCH * NHEADS), blk, 0, stream>>>(QpH, QpL, KpH, KpL, attn);
    softmax_rows<<<NBATCH * NHEADS * SEQ, blk, 0, stream>>>(attn);

    // Context (attn split on the fly; V pre-split), then output projection.
    attn_context<<<dim3(16, 1, NBATCH * NHEADS), blk, 0, stream>>>(attn, VtH, VtL, cxH, cxL);
    proj_gemm<<<512, blk, 0, stream>>>(cxH, cxL, WoH, WoL, bo, out, nullptr, nullptr, 2);
}
